// MultiHeadAttention_89824946029029
// MI455X (gfx1250) — matrix operand order repaired and checked by reading, not compiled, not run on hardware
//
#include <hip/hip_runtime.h>
#include <hip/hip_bf16.h>

typedef __attribute__((ext_vector_type(16))) __bf16 v16bf;
typedef __attribute__((ext_vector_type(8)))  __bf16 v8bf;
typedef __attribute__((ext_vector_type(8)))  float  v8f;
typedef __attribute__((ext_vector_type(4)))  unsigned int v4u;
typedef __attribute__((ext_vector_type(8)))  int    v8i;
typedef __attribute__((ext_vector_type(4)))  int    v4i;

#define D_MODEL 512
#define NUM_HEADS 8
#define HEAD 64
#define BATCH 4
#define SEQ 2048
#define ROWS (BATCH*SEQ)   // 8192

// ---- bf16 helpers (round-to-nearest-even) ------------------------------
__device__ __forceinline__ unsigned short f2bf(float f) {
    unsigned u = __float_as_uint(f);
    u += 0x7FFFu + ((u >> 16) & 1u);
    return (unsigned short)(u >> 16);
}

// Assemble a 16-element bf16 fragment from two contiguous 16B segments.
__device__ __forceinline__ v16bf ldfrag(const unsigned short* p0,
                                        const unsigned short* p1) {
    v8bf lo = *reinterpret_cast<const v8bf*>(p0);
    v8bf hi = *reinterpret_cast<const v8bf*>(p1);
    return __builtin_shufflevector(lo, hi,
        0,1,2,3,4,5,6,7,8,9,10,11,12,13,14,15);
}

// ---- TDM: DMA a 64(dh) x 64(key) bf16 tile (row stride SEQ) into LDS ----
// LDS layout produced: [dh][key] with 72-half row stride via pad fields
// (pad_interval=4 -> pad every 32 DWORDs = 128B; pad_amount=3 -> 4 DWORDs = 16B).
__device__ __forceinline__ void tdm_load_vt(const unsigned short* gsrc,
                                            unsigned lds_addr) {
    unsigned long long ga = (unsigned long long)(uintptr_t)gsrc;
    v4u g0 = { 1u,                               // count=1 (valid descriptor)
               lds_addr,                         // lds_addr (bytes)
               (unsigned)ga,                     // global_addr[31:0]
               (unsigned)((ga >> 32) & 0x01FFFFFFu) | (2u << 30) }; // ga[56:32] | type=2
    v8i g1 = { (int)((1u << 16)   // data_size = 2 bytes
                   | (1u << 20)   // pad_enable
                   | (4u << 22)   // pad_interval: 32 DWORDs
                   | (3u << 25)), // pad_amount: 4 DWORDs
               (int)(64u << 16),  // tensor_dim0 = 64  (bits 79:48 -> [31:16])
               (int)(64u << 16),  // tensor_dim1 = 64  (bits 111:80 -> [31:16])
               (int)(64u << 16),  // tile_dim0  = 64  (bits 127:112)
               64,                // tile_dim1  = 64  (bits 143:128)
               (int)SEQ,          // tensor_dim0_stride = 2048 (bits 207:160 lo)
               0, 0 };
    v4i gz4 = { 0, 0, 0, 0 };
    v8i gz8 = { 0, 0, 0, 0, 0, 0, 0, 0 };
    __builtin_amdgcn_tensor_load_to_lds(g0, g1, gz4, gz4, gz8, 0);
}

// ---- fp32 -> bf16 cast (vectorized x4) ---------------------------------
__global__ void cast_bf16(const float* __restrict__ src,
                          unsigned short* __restrict__ dst, int n) {
    int i = (blockIdx.x * blockDim.x + threadIdx.x) * 4;
    if (i < n) {
        float4 v = *reinterpret_cast<const float4*>(src + i);
        ushort4 o;
        o.x = f2bf(v.x); o.y = f2bf(v.y); o.z = f2bf(v.z); o.w = f2bf(v.w);
        *reinterpret_cast<ushort4*>(dst + i) = o;
    }
}

// ---- GEMM: Y[s,e] = sum_d X[s,d] * W[e,d] + bias[e] --------------------
// MODE 0: bf16 -> head-major [b, h, s, 64]      (Q, K projections)
// MODE 1: f32  -> row-major  [g, 512]           (output projection)
// MODE 2: bf16 -> transposed head-major [b, h, 64, s]  (V projection, for TDM)
template <int MODE>
__global__ __launch_bounds__(256)
void proj_gemm(const unsigned short* __restrict__ X,
               const unsigned short* __restrict__ W,
               const float* __restrict__ bias,
               unsigned short* __restrict__ outH,
               float* __restrict__ outF) {
    const int lane = threadIdx.x & 31;
    const int wave = threadIdx.x >> 5;
    const int n16  = lane & 15;
    const int hi   = lane >> 4;
    const int rbase = blockIdx.y * 128 + wave * 16;
    const int cbase = blockIdx.x * 64;

    v8f acc[4] = {};
    const unsigned short* Xrow = X + (size_t)(rbase + n16) * D_MODEL;

    #pragma unroll 4
    for (int k0 = 0; k0 < D_MODEL; k0 += 32) {
        // A fragment: lane = row, K segments at (8*hi) and (16 + 8*hi)
        v16bf a = ldfrag(Xrow + k0 + hi * 8, Xrow + k0 + 16 + hi * 8);
        #pragma unroll
        for (int nt = 0; nt < 4; nt++) {
            // B fragment: lane = column e, K segments at (8*hi) and (16 + 8*hi)
            const unsigned short* Wrow =
                W + (size_t)(cbase + nt * 16 + n16) * D_MODEL + k0 + hi * 8;
            v16bf b = ldfrag(Wrow, Wrow + 16);
            acc[nt] = __builtin_amdgcn_wmma_f32_16x16x32_bf16(
                false, a, false, b, (short)0, acc[nt], false, false);
        }
    }

    #pragma unroll
    for (int nt = 0; nt < 4; nt++) {
        const int e = cbase + nt * 16 + n16;
        const float bv = bias[e];
        #pragma unroll
        for (int r = 0; r < 8; r++) {
            const int g = rbase + r + hi * 8;     // global row (C/D layout)
            const float val = acc[nt][r] + bv;
            const int bidx = g >> 11, s = g & (SEQ - 1);
            const int h = e >> 6, dh = e & 63;
            if constexpr (MODE == 0) {
                outH[((size_t)(bidx * NUM_HEADS + h) * SEQ + s) * HEAD + dh] =
                    f2bf(val);
            } else if constexpr (MODE == 2) {
                outH[((size_t)(bidx * NUM_HEADS + h) * HEAD + dh) * SEQ + s] =
                    f2bf(val);
            } else {
                outF[(size_t)g * D_MODEL + e] = val;
            }
        }
    }
}

// ---- Flash attention: one wave per 16 query rows -----------------------
// Block 128 thr (4 waves) covers 64 q rows. Grid (2048/64, B*H).
// V^T chunks are DMA'd by the Tensor Data Mover into double-buffered LDS,
// overlapped with the WMMA + softmax of the previous chunk.
__global__ __launch_bounds__(128)
void flash_attn(const unsigned short* __restrict__ Qp,
                const unsigned short* __restrict__ Kp,
                const unsigned short* __restrict__ Vt_g,   // [b,h,64,2048]
                const int* __restrict__ mask,
                unsigned short* __restrict__ ctx) {
    __shared__ unsigned short Vt[2][64 * 72];     // V^T chunk [dh][key], padded
    __shared__ unsigned short Pl[4][16 * 72];     // per-wave P tile [row][key]

    const int lane = threadIdx.x & 31;
    const int wave = threadIdx.x >> 5;
    const int n16  = lane & 15;
    const int hi   = lane >> 4;
    const int bh   = blockIdx.y;                  // 0..31
    const int bidx = bh >> 3, h = bh & 7;
    const int q0   = blockIdx.x * 64 + wave * 16;

    const unsigned short* vbase = Vt_g + (size_t)bh * HEAD * SEQ;

    // Kick off DMA of the first V^T chunk (wave 0 only; TDM ignores EXEC).
    if (wave == 0) {
        tdm_load_vt(vbase, (unsigned)(uintptr_t)&Vt[0][0]);
    }

    // Q fragments (dh split into two K=32 steps), loaded once
    const unsigned short* Qrow = Qp + (size_t)(bh * SEQ + q0 + n16) * HEAD;
    const v16bf qa0 = ldfrag(Qrow + hi * 8,      Qrow + 16 + hi * 8);
    const v16bf qa1 = ldfrag(Qrow + 32 + hi * 8, Qrow + 48 + hi * 8);

    float mrun[8], lrun[8];
    v8f Oacc[4] = {};
    #pragma unroll
    for (int r = 0; r < 8; r++) { mrun[r] = -INFINITY; lrun[r] = 0.f; }

    const int* mrow = mask + (size_t)bidx * SEQ * SEQ;

    for (int kc = 0; kc < SEQ; kc += 64) {
        const int p = (kc >> 6) & 1;
        // DMA of buffer p (issued last iteration / before loop) must land.
        if (wave == 0) __builtin_amdgcn_s_wait_tensorcnt(0);
        __syncthreads();   // buf p ready; everyone done reading buf 1-p
        // Prefetch next chunk into the other buffer, overlapped with compute.
        if (wave == 0 && kc + 64 < SEQ) {
            tdm_load_vt(vbase + (kc + 64),
                        (unsigned)(uintptr_t)&Vt[1 - p][0]);
        }

        // ---- scores S = (Q K^T) / 8, masked ----
        v8f smat[4];
        #pragma unroll
        for (int nt = 0; nt < 4; nt++) {
            v8f s = {};
            const unsigned short* Krow =
                Kp + (size_t)(bh * SEQ + kc + nt * 16 + n16) * HEAD + hi * 8;
            v16bf kb0 = ldfrag(Krow,      Krow + 16);
            v16bf kb1 = ldfrag(Krow + 32, Krow + 48);
            s = __builtin_amdgcn_wmma_f32_16x16x32_bf16(
                false, qa0, false, kb0, (short)0, s, false, false);
            s = __builtin_amdgcn_wmma_f32_16x16x32_bf16(
                false, qa1, false, kb1, (short)0, s, false, false);
            const int kcol = kc + nt * 16 + n16;
            #pragma unroll
            for (int r = 0; r < 8; r++) {
                float sv = s[r] * 0.125f;             // 1/sqrt(64)
                int mv = mrow[(size_t)(q0 + r + hi * 8) * SEQ + kcol];
                smat[nt][r] = (mv == 0) ? -INFINITY : sv;
            }
        }

        // ---- online softmax statistics (rows live across 16-lane groups) ----
        float corr[8], psum[8];
        #pragma unroll
        for (int r = 0; r < 8; r++) {
            float m = fmaxf(fmaxf(smat[0][r], smat[1][r]),
                            fmaxf(smat[2][r], smat[3][r]));
            #pragma unroll
            for (int off = 8; off > 0; off >>= 1)
                m = fmaxf(m, __shfl_xor(m, off, 32)); // stays in 16-lane group
            float mnew = fmaxf(mrun[r], m);
            corr[r] = (mrun[r] == -INFINITY) ? 0.f : __expf(mrun[r] - mnew);
            mrun[r] = mnew;
            psum[r] = 0.f;
        }

        // ---- P = exp(S - m), transpose via per-wave LDS into A layout ----
        #pragma unroll
        for (int nt = 0; nt < 4; nt++) {
            #pragma unroll
            for (int r = 0; r < 8; r++) {
                float sv = smat[nt][r];
                float pv = (sv == -INFINITY) ? 0.f : __expf(sv - mrun[r]);
                psum[r] += pv;
                Pl[wave][(r + hi * 8) * 72 + nt * 16 + n16] = f2bf(pv);
            }
        }
        #pragma unroll
        for (int r = 0; r < 8; r++) {
            float v = psum[r];
            #pragma unroll
            for (int off = 8; off > 0; off >>= 1)
                v += __shfl_xor(v, off, 32);
            lrun[r] = lrun[r] * corr[r] + v;
        }

        // ---- rescale running output ----
        #pragma unroll
        for (int nt = 0; nt < 4; nt++)
            #pragma unroll
            for (int r = 0; r < 8; r++)
                Oacc[nt][r] *= corr[r];

        // ---- O += P · V ----
        #pragma unroll
        for (int ks = 0; ks < 2; ks++) {
            const unsigned short* prow =
                &Pl[wave][n16 * 72 + ks * 32 + hi * 8];
            v16bf pa = ldfrag(prow, prow + 16);       // A frag from LDS
            #pragma unroll
            for (int nt = 0; nt < 4; nt++) {
                const unsigned short* vrow =
                    &Vt[p][(nt * 16 + n16) * 72 + ks * 32 + hi * 8];
                v16bf vb = ldfrag(vrow, vrow + 16);   // B frag from LDS (V^T)
                Oacc[nt] = __builtin_amdgcn_wmma_f32_16x16x32_bf16(
                    false, pa, false, vb, (short)0, Oacc[nt], false, false);
            }
        }
    }

    // ---- epilogue: ctx[b, s, h*64+dh] = O / l ----
    #pragma unroll
    for (int nt = 0; nt < 4; nt++) {
        #pragma unroll
        for (int r = 0; r < 8; r++) {
            const int qrow = q0 + r + hi * 8;
            float o = Oacc[nt][r] / lrun[r];
            ctx[((size_t)(bidx * SEQ + qrow)) * D_MODEL + h * HEAD + nt * 16 + n16] =
                f2bf(o);
        }
    }
}

// ---- host launch --------------------------------------------------------
extern "C" void kernel_launch(void* const* d_in, const int* in_sizes, int n_in,
                              void* d_out, int out_size, void* d_ws, size_t ws_size,
                              hipStream_t stream) {
    const float* query = (const float*)d_in[0];
    const float* key   = (const float*)d_in[1];
    const float* value = (const float*)d_in[2];
    const int*   mask  = (const int*)d_in[3];
    const float* Wq = (const float*)d_in[4];
    const float* bq = (const float*)d_in[5];
    const float* Wk = (const float*)d_in[6];
    const float* bk = (const float*)d_in[7];
    const float* Wv = (const float*)d_in[8];
    const float* bv = (const float*)d_in[9];
    const float* Wo = (const float*)d_in[10];
    const float* bo = (const float*)d_in[11];
    (void)in_sizes; (void)n_in; (void)out_size; (void)ws_size;

    char* ws = (char*)d_ws;
    const size_t NX = (size_t)ROWS * D_MODEL;        // 4,194,304 elems
    const size_t NW = (size_t)D_MODEL * D_MODEL;     //   262,144 elems
    unsigned short* Xq   = (unsigned short*)(ws);
    unsigned short* Xk   = (unsigned short*)(ws + 2 * NX);
    unsigned short* Xv   = (unsigned short*)(ws + 4 * NX);
    unsigned short* wq16 = (unsigned short*)(ws + 6 * NX);
    unsigned short* wk16 = (unsigned short*)(ws + 6 * NX + 2 * NW);
    unsigned short* wv16 = (unsigned short*)(ws + 6 * NX + 4 * NW);
    unsigned short* wo16 = (unsigned short*)(ws + 6 * NX + 6 * NW);
    unsigned short* Qp   = (unsigned short*)(ws + 6 * NX + 8 * NW);
    unsigned short* Kp   = (unsigned short*)(ws + 8 * NX + 8 * NW);
    unsigned short* Vtg  = (unsigned short*)(ws + 10 * NX + 8 * NW);
    unsigned short* ctxb = (unsigned short*)(ws + 12 * NX + 8 * NW);

    // 1. fp32 -> bf16 casts
    cast_bf16<<<(int)(NX / 1024), 256, 0, stream>>>(query, Xq, (int)NX);
    cast_bf16<<<(int)(NX / 1024), 256, 0, stream>>>(key,   Xk, (int)NX);
    cast_bf16<<<(int)(NX / 1024), 256, 0, stream>>>(value, Xv, (int)NX);
    cast_bf16<<<(int)(NW / 1024), 256, 0, stream>>>(Wq, wq16, (int)NW);
    cast_bf16<<<(int)(NW / 1024), 256, 0, stream>>>(Wk, wk16, (int)NW);
    cast_bf16<<<(int)(NW / 1024), 256, 0, stream>>>(Wv, wv16, (int)NW);
    cast_bf16<<<(int)(NW / 1024), 256, 0, stream>>>(Wo, wo16, (int)NW);

    // 2. Q/K head-major, V transposed head-major (TDM-friendly)
    dim3 pgrid(D_MODEL / 64, ROWS / 128);
    proj_gemm<0><<<pgrid, 256, 0, stream>>>(Xq, wq16, bq, Qp, nullptr);
    proj_gemm<0><<<pgrid, 256, 0, stream>>>(Xk, wk16, bk, Kp, nullptr);
    proj_gemm<2><<<pgrid, 256, 0, stream>>>(Xv, wv16, bv, Vtg, nullptr);

    // 3. flash attention -> ctx [b, s, 512] bf16
    flash_attn<<<dim3(SEQ / 64, BATCH * NUM_HEADS), 128, 0, stream>>>(
        Qp, Kp, Vtg, mask, ctxb);

    // 4. output projection -> fp32 d_out
    proj_gemm<1><<<pgrid, 256, 0, stream>>>(ctxb, wo16, bo, nullptr,
                                            (float*)d_out);
}
